// MonolithicSSMLayer_62302795596482
// MI455X (gfx1250) — matrix operand order repaired and checked by reading, not compiled, not run on hardware
//
#include <hip/hip_runtime.h>

// ---------------------------------------------------------------------------
// MonolithicSSMLayer on MI455X (gfx1250, wave32):
//   Bx = x @ B_w^T + B_b          (WMMA bf16/f32, async-LDS pipelined)
//   h  = scan(A, Bx)              (chunked: local scans + carry combine)
//   y  = h @ C_w^T + C_b          (WMMA, carry-correction fused in staging)
// ---------------------------------------------------------------------------

typedef __attribute__((ext_vector_type(16))) __bf16    v16bf;
typedef __attribute__((ext_vector_type(2)))  __bf16    v2bf;
typedef __attribute__((ext_vector_type(2)))  float     v2f;
typedef __attribute__((ext_vector_type(8)))  float     v8f;
typedef __attribute__((ext_vector_type(8)))  unsigned  v8u;

#define D_MODEL 1024
#define N_STATE 1024
#define M_TOK   (8 * 2048)      // B*T tokens
#define CHUNK   128             // scan chunk length
#define NCHUNK  (M_TOK / CHUNK) // 128 global chunks (= B*16)

#define ROWF    36              // f32 LDS row stride (144 B, 16B aligned)
#define ROWP    18              // bf16 LDS row stride in uint pairs (72 B)

// ----- helpers --------------------------------------------------------------

__device__ __forceinline__ unsigned packbf(float lo, float hi) {
    v2f f = {lo, hi};
    v2bf b = __builtin_convertvector(f, v2bf);   // single v_cvt_pk_bf16_f32
    return __builtin_bit_cast(unsigned, b);
}

__device__ __forceinline__ unsigned lds_addr(const void* p) {
    return (unsigned)(reinterpret_cast<uintptr_t>(p));   // low 32b = LDS offset
}

// Async-copy one 128x32 f32 tile (global row stride K floats) into a padded
// LDS tile. 1024 16B chunks, 4 per thread -> 4 async ops per lane per tile.
__device__ __forceinline__ void issue_tile(const float* __restrict__ g0, int K,
                                           float* l0, int tid) {
    #pragma unroll
    for (int i = 0; i < 4; ++i) {
        const int chunk = tid + i * 256;
        const int row = chunk >> 3, c = chunk & 7;
        const void* g = (const char*)(g0 + (size_t)row * K) + c * 16;
        const unsigned l = lds_addr((const char*)(l0 + row * ROWF) + c * 16);
        asm volatile("global_load_async_to_lds_b128 %0, %1, off"
                     :: "v"(l), "v"(g) : "memory");
    }
}

// ----- kernel 1: A = sigmoid(log_A); L2A = log2(A); A128 = A^128 ------------

__global__ __launch_bounds__(256) void precompute_kernel(
    const float* __restrict__ logA,
    float* __restrict__ A, float* __restrict__ L2A, float* __restrict__ A128) {
    int n = blockIdx.x * blockDim.x + threadIdx.x;   // 0..1023
    float a = 1.0f / (1.0f + __expf(-logA[n]));
    A[n] = a;
    L2A[n] = __log2f(a);
    float p = a;
    #pragma unroll
    for (int i = 0; i < 7; ++i) p = p * p;           // a^128
    A128[n] = p;
}

// ----- kernels 2/5: NT GEMM  out[m,n] = sum_k Amat[m,k]*Bmat[n,k] + bias[n] -
// Block: 256 threads = 8 waves (2 x 4). Block tile 128x128, wave tile 64x32.
// Per k-step: async f32 tiles -> LDS, one conversion pass f32->bf16 (fusing
// the scan-carry correction h += A^(row+1)*E when CORRECT), then a pure-LDS
// bf16 fragment gather feeding 8 WMMAs, overlapped with the next async copy.

struct GemmSmem {
    float    fA[128 * ROWF];   // async landing pads (f32)
    float    fB[128 * ROWF];
    unsigned bA[128 * ROWP];   // packed bf16 tiles (compute buffers)
    unsigned bB[128 * ROWP];
    float    E[32];            // carry slice for current k-step
    float    L[32];            // log2(A) slice for current k-step
};

template <bool CORRECT>
__global__ __launch_bounds__(256) void gemm_nt_wmma(
    const float* __restrict__ Amat,   // [M, K] f32
    const float* __restrict__ Bmat,   // [N, K] f32 (weights, row-major [out,in])
    const float* __restrict__ bias,   // [N]
    float* __restrict__ out,          // [M, N]
    const float* __restrict__ L2A,    // [K]          (CORRECT only)
    const float* __restrict__ Ecar,   // [NCHUNK, K]  (CORRECT only)
    int K, int N) {

    __shared__ GemmSmem sm;

    const int tid = threadIdx.x;
    const int m0 = blockIdx.y * 128;
    const int n0 = blockIdx.x * 128;

    const int lane   = tid & 31;
    const int wid    = tid >> 5;             // 0..7
    const int wave_m = wid >> 2;             // 0..1  -> 64-row slab
    const int wave_n = wid & 3;              // 0..3  -> 32-col slab
    const bool hi    = (lane & 16) != 0;
    const int  l15   = lane & 15;

    // A-fragment k-pair indices: contiguous k order (k = 16*hi + s), matching B
    int apair[8];
    #pragma unroll
    for (int i = 0; i < 8; ++i)
        apair[i] = i + (hi ? 8 : 0);

    const float* Asrc = Amat + (size_t)m0 * K;
    const float* Bsrc = Bmat + (size_t)n0 * K;
    const float* Erow = CORRECT ? (Ecar + (size_t)(m0 >> 7) * K) : nullptr;

    // conversion-pass mapping: thread -> (row, 16-wide k-half)
    const int crow = tid >> 1;
    const int ckh  = (tid & 1) * 16;

    v8f acc[4][2];
    #pragma unroll
    for (int fm = 0; fm < 4; ++fm)
        #pragma unroll
        for (int fn = 0; fn < 2; ++fn)
            acc[fm][fn] = (v8f){};

    // ---- prologue: prefetch k-step 0 ----
    issue_tile(Asrc, K, sm.fA, tid);
    issue_tile(Bsrc, K, sm.fB, tid);
    if constexpr (CORRECT) {
        if (tid < 32)                 sm.E[tid]      = Erow[tid];
        else if (tid < 64)            sm.L[tid - 32] = L2A[tid - 32];
    }

    for (int kt = 0; kt < K; kt += 32) {
        asm volatile("s_wait_asynccnt 0" ::: "memory");
        __syncthreads();                       // f32 tiles + E/L resident

        // ---- conversion pass: f32 -> packed bf16 (fuse correction on A) ----
        {
            const float4* srcA = reinterpret_cast<const float4*>(sm.fA + crow * ROWF + ckh);
            const float4* srcB = reinterpret_cast<const float4*>(sm.fB + crow * ROWF + ckh);
            uint2* dstA = reinterpret_cast<uint2*>(sm.bA + crow * ROWP + (ckh >> 1));
            uint2* dstB = reinterpret_cast<uint2*>(sm.bB + crow * ROWP + (ckh >> 1));
            const float rp1 = (float)(crow + 1);

            float4 ha[4], wb[4];
            #pragma unroll
            for (int j = 0; j < 4; ++j) { ha[j] = srcA[j]; wb[j] = srcB[j]; }
            if constexpr (CORRECT) {
                #pragma unroll
                for (int j = 0; j < 4; ++j) {
                    const float4 e4 = reinterpret_cast<const float4*>(sm.E + ckh)[j];
                    const float4 l4 = reinterpret_cast<const float4*>(sm.L + ckh)[j];
                    ha[j].x = fmaf(exp2f(rp1 * l4.x), e4.x, ha[j].x);  // + A^(row+1)*E
                    ha[j].y = fmaf(exp2f(rp1 * l4.y), e4.y, ha[j].y);
                    ha[j].z = fmaf(exp2f(rp1 * l4.z), e4.z, ha[j].z);
                    ha[j].w = fmaf(exp2f(rp1 * l4.w), e4.w, ha[j].w);
                }
            }
            #pragma unroll
            for (int j = 0; j < 4; ++j) {
                uint2 ua; ua.x = packbf(ha[j].x, ha[j].y); ua.y = packbf(ha[j].z, ha[j].w);
                dstA[j] = ua;
                uint2 ub; ub.x = packbf(wb[j].x, wb[j].y); ub.y = packbf(wb[j].z, wb[j].w);
                dstB[j] = ub;
            }
        }
        __syncthreads();                       // bf16 tiles ready; f32 free

        // ---- prefetch next k-step into the (now free) f32 tiles ----
        if (kt + 32 < K) {
            issue_tile(Asrc + kt + 32, K, sm.fA, tid);
            issue_tile(Bsrc + kt + 32, K, sm.fB, tid);
            if constexpr (CORRECT) {
                if (tid < 32)      sm.E[tid]      = Erow[kt + 32 + tid];
                else if (tid < 64) sm.L[tid - 32] = L2A[kt + 32 + tid - 32];
            }
        }

        // ---- gather B fragments (32x16: col = lane&15, K contiguous) ----
        v16bf bfrag[2];
        #pragma unroll
        for (int fn = 0; fn < 2; ++fn) {
            const int bcol = wave_n * 32 + fn * 16 + l15;
            v8u bu;
            #pragma unroll
            for (int i = 0; i < 8; ++i)
                bu[i] = sm.bB[bcol * ROWP + (hi ? 8 : 0) + i];
            bfrag[fn] = __builtin_bit_cast(v16bf, bu);
        }
        // ---- gather A fragments (16x32: row = lane&15, K contiguous) + 8 WMMAs ----
        #pragma unroll
        for (int fm = 0; fm < 4; ++fm) {
            const int arow = wave_m * 64 + fm * 16 + l15;
            v8u au;
            #pragma unroll
            for (int i = 0; i < 8; ++i)
                au[i] = sm.bA[arow * ROWP + apair[i]];
            v16bf afrag = __builtin_bit_cast(v16bf, au);
            #pragma unroll
            for (int fn = 0; fn < 2; ++fn) {
                acc[fm][fn] = __builtin_amdgcn_wmma_f32_16x16x32_bf16(
                    false, afrag, false, bfrag[fn],
                    (short)0, acc[fm][fn], false, false);
            }
        }
    }

    // ---- epilogue: D layout (row = r + 8*hi, col = lane&15) per VGPR r ----
    #pragma unroll
    for (int fn = 0; fn < 2; ++fn) {
        const int col = n0 + wave_n * 32 + fn * 16 + l15;
        const float bv = bias[col];
        #pragma unroll
        for (int fm = 0; fm < 4; ++fm) {
            #pragma unroll
            for (int r = 0; r < 8; ++r) {
                const int mrow = m0 + wave_m * 64 + fm * 16 + r + (hi ? 8 : 0);
                out[(size_t)mrow * N + col] = acc[fm][fn][r] + bv;
            }
        }
    }
}

// ----- kernel 3: local scans over 128-step chunks (in place), chunk sums S --
// One thread handles 4 adjacent states (float4 streaming).

__global__ __launch_bounds__(256) void scan_local_kernel(
    const float* __restrict__ A, float* __restrict__ Bx, float* __restrict__ S) {
    const int id = blockIdx.x * blockDim.x + threadIdx.x;  // NCHUNK*N_STATE/4
    const int n4 = (id & (N_STATE / 4 - 1)) * 4;
    const int cg = id >> 8;                                // global chunk (b*16+c)
    const float4 a = *reinterpret_cast<const float4*>(A + n4);
    float4* base = reinterpret_cast<float4*>(Bx + (size_t)cg * CHUNK * N_STATE + n4);
    float4 h = {0.f, 0.f, 0.f, 0.f};
    #pragma unroll 4
    for (int t = 0; t < CHUNK; ++t) {
        float4* p = base + t * (N_STATE / 4);
        const float4 v = *p;
        h.x = fmaf(a.x, h.x, v.x);
        h.y = fmaf(a.y, h.y, v.y);
        h.z = fmaf(a.z, h.z, v.z);
        h.w = fmaf(a.w, h.w, v.w);
        *p = h;                                            // now h_local
    }
    *reinterpret_cast<float4*>(S + (size_t)cg * N_STATE + n4) = h;
}

// ----- kernel 4: exclusive carry combine per (batch, n), 4 states/thread ----

__global__ __launch_bounds__(256) void scan_carry_kernel(
    const float* __restrict__ A128, const float* __restrict__ S,
    float* __restrict__ E) {
    const int id = blockIdx.x * blockDim.x + threadIdx.x;  // 8*N_STATE/4
    const int n4 = (id & (N_STATE / 4 - 1)) * 4;
    const int b  = id >> 8;
    const float4 a = *reinterpret_cast<const float4*>(A128 + n4);
    float4 carry = {0.f, 0.f, 0.f, 0.f};
    #pragma unroll
    for (int c = 0; c < 16; ++c) {
        const size_t i = (size_t)(b * 16 + c) * N_STATE + n4;
        *reinterpret_cast<float4*>(E + i) = carry;
        const float4 s = *reinterpret_cast<const float4*>(S + i);
        carry.x = fmaf(a.x, carry.x, s.x);
        carry.y = fmaf(a.y, carry.y, s.y);
        carry.z = fmaf(a.z, carry.z, s.z);
        carry.w = fmaf(a.w, carry.w, s.w);
    }
}

// ----- launch ---------------------------------------------------------------

extern "C" void kernel_launch(void* const* d_in, const int* in_sizes, int n_in,
                              void* d_out, int out_size, void* d_ws, size_t ws_size,
                              hipStream_t stream) {
    const float* x    = (const float*)d_in[0];
    const float* logA = (const float*)d_in[1];
    const float* Bw   = (const float*)d_in[2];
    const float* Bb   = (const float*)d_in[3];
    const float* Cw   = (const float*)d_in[4];
    const float* Cb   = (const float*)d_in[5];
    float* y  = (float*)d_out;
    float* ws = (float*)d_ws;

    // workspace layout (floats)
    float* Bx   = ws;                                   // M_TOK*N_STATE (64 MB)
    float* A    = Bx   + (size_t)M_TOK * N_STATE;       // 1024
    float* A128 = A    + N_STATE;                       // 1024
    float* L2A  = A128 + N_STATE;                       // 1024
    float* S    = L2A  + N_STATE;                       // NCHUNK*N_STATE
    float* E    = S    + (size_t)NCHUNK * N_STATE;      // NCHUNK*N_STATE

    precompute_kernel<<<N_STATE / 256, 256, 0, stream>>>(logA, A, L2A, A128);

    dim3 gGemm(N_STATE / 128, M_TOK / 128);
    gemm_nt_wmma<false><<<gGemm, 256, 0, stream>>>(
        x, Bw, Bb, Bx, nullptr, nullptr, D_MODEL, N_STATE);

    scan_local_kernel<<<(NCHUNK * N_STATE / 4) / 256, 256, 0, stream>>>(A, Bx, S);
    scan_carry_kernel<<<(8 * N_STATE / 4) / 256, 256, 0, stream>>>(A128, S, E);

    gemm_nt_wmma<true><<<gGemm, 256, 0, stream>>>(
        Bx, Cw, Cb, y, L2A, E, N_STATE, D_MODEL);
}
